// neural_network_equivariant_28965259444729
// MI455X (gfx1250) — matrix operand order repaired and checked by reading, not compiled, not run on hardware
//
#include <hip/hip_runtime.h>
#include <hip/hip_bf16.h>
#include <math.h>

// ---------------- constants from the reference ----------------
#define NN_ED 32
#define NN_NB 8
#define NN_MS 16
#define NN_MV 8
#define NN_CS 24      // MS + MV
#define NN_L  2
#define NN_RH 64
#define NN_WNUM 640   // MS*2*MV + MV*2*CS = 256 + 384
#define NN_NA 256
#define PI_F 3.14159265358979f

typedef __attribute__((ext_vector_type(16))) _Float16 v16h;
typedef __attribute__((ext_vector_type(8)))  _Float16 v8h;
typedef __attribute__((ext_vector_type(8)))  float    v8f;

static __device__ inline v8f wmma16x16x32(v16h A, v16h B, v8f C) {
  return __builtin_amdgcn_wmma_f32_16x16x32_f16(false, A, false, B, (short)0, C, false, false);
}
// fast silu/sigmoid: v_exp + v_rcp, no precise-division expansion
static __device__ inline float silu_f(float x) {
  return x * __builtin_amdgcn_rcpf(1.f + __expf(-x));
}
static __device__ inline float sigm_f(float x) {
  return __builtin_amdgcn_rcpf(1.f + __expf(-x));
}

// ------------- weight pre-shuffle into WMMA B-tile layout (f16) -------------
// B layout (16-bit, 32x16 tile, wave32): lane l holds column n = l%16;
// half index j holds K = j + 8*(l/16) for j<8, 16 + 8*(l/16) + (j-8) for j>=8 (+32*chunk for the second K-chunk).
// MLP normalizations (1/sqrt(32), 1/sqrt(RH)) are folded into the weights.
__global__ void k_convw1(const float* __restrict__ fc_w1, _Float16* __restrict__ W1B) {
  int idx = blockIdx.x * blockDim.x + threadIdx.x;
  if (idx >= NN_L * 4 * 32 * 16) return;
  int j = idx & 15, lane = (idx >> 4) & 31, t = (idx >> 9) & 3, layer = idx >> 11;
  int k = (j < 8) ? j + ((lane < 16) ? 0 : 8) : 16 + ((lane < 16) ? 0 : 8) + (j - 8); // padded K (16..31 -> zero)
  int n = t * 16 + (lane & 15);
  float v = (k < 16) ? fc_w1[(layer * 16 + k) * NN_RH + n] * 0.17677670f : 0.f; // 1/sqrt(32)
  W1B[idx] = (_Float16)v;
}

__global__ void k_convw2(const float* __restrict__ fc_w2, _Float16* __restrict__ W2B) {
  int idx = blockIdx.x * blockDim.x + threadIdx.x;
  if (idx >= NN_L * 40 * 2 * 32 * 16) return;
  int j = idx & 15, lane = (idx >> 4) & 31, chunk = (idx >> 9) & 1;
  int rest = idx >> 10, t = rest % 40, layer = rest / 40;
  int k = chunk * 32 + ((j < 8) ? j + ((lane < 16) ? 0 : 8) : 16 + ((lane < 16) ? 0 : 8) + (j - 8));
  int n = t * 16 + (lane & 15);
  W2B[idx] = (_Float16)(fc_w2[(layer * NN_RH + k) * NN_WNUM + n] * 0.125f);     // 1/sqrt(64)
}

// ------------- init: embedding, uplift yv = x . K, zero scalars -------------
__global__ void k_init(const float* __restrict__ x, const int* __restrict__ node_attr,
                       const float* __restrict__ emb, const float* __restrict__ Kc,
                       float* __restrict__ a, float* __restrict__ ys, float* __restrict__ ys_old,
                       float* __restrict__ yv, float* __restrict__ yv_old,
                       float* __restrict__ xcur, int Nn) {
  int n = blockIdx.x * blockDim.x + threadIdx.x;
  if (n >= Nn) return;
  int at = node_attr[n];
#pragma unroll
  for (int e = 0; e < NN_ED; ++e) a[n * NN_ED + e] = emb[at * NN_ED + e];
#pragma unroll
  for (int j = 0; j < 6; ++j) xcur[n * 6 + j] = x[n * 6 + j];
#pragma unroll
  for (int k = 0; k < NN_MV; ++k)
#pragma unroll
    for (int d = 0; d < 3; ++d) {
      float v = x[n * 6 + d] * Kc[k] + x[n * 6 + 3 + d] * Kc[NN_MV + k];
      yv[n * 24 + k * 3 + d] = v;
      yv_old[n * 24 + k * 3 + d] = v;
    }
#pragma unroll
  for (int j = 0; j < NN_MS; ++j) { ys[n * NN_MS + j] = 0.f; ys_old[n * NN_MS + j] = 0.f; }
}

// ------------- node linears: scs, scv, zs, zv (wave per node) -------------
__global__ __launch_bounds__(256) void k_nodelin(
    const float* __restrict__ a, const float* __restrict__ ys, const float* __restrict__ yv,
    const float* __restrict__ sc_s, const float* __restrict__ sc_v,
    const float* __restrict__ lin1_s, const float* __restrict__ lin1_v,
    float* __restrict__ scs, float* __restrict__ scv,
    float* __restrict__ zs, float* __restrict__ zv, int layer, int Nn) {
  __shared__ float aL[8][32], ysL[8][16], yvL[8][24];
  int w = threadIdx.x >> 5, l = threadIdx.x & 31;
  int n = blockIdx.x * 8 + w;
  bool act = n < Nn;
  if (act) {
    aL[w][l] = a[n * 32 + l];
    if (l < 16) ysL[w][l] = ys[n * 16 + l];
    if (l < 24) yvL[w][l] = yv[n * 24 + l];
  }
  __syncthreads();
  if (!act) return;
  for (int idx = l; idx < 88; idx += 32) {
    if (idx < 24) {                       // scs [CS]
      int o = idx; float acc = 0.f;
      for (int i = 0; i < 16; ++i) {
        float yi = ysL[w][i];
        const float* row = sc_s + (size_t)((layer * 16 + i) * 32) * 24 + o;
#pragma unroll 8
        for (int e = 0; e < 32; ++e) acc += yi * aL[w][e] * row[e * 24];
      }
      scs[n * 24 + o] = acc * 0.044194174f;                 // 1/sqrt(512)
    } else if (idx < 48) {                // scv [MV,3]
      int j = idx - 24, o = j / 3, d = j % 3; float acc = 0.f;
      for (int i = 0; i < 8; ++i) {
        float yi = yvL[w][i * 3 + d];
        const float* row = sc_v + (size_t)((layer * 8 + i) * 32) * 8 + o;
#pragma unroll 8
        for (int e = 0; e < 32; ++e) acc += yi * aL[w][e] * row[e * 8];
      }
      scv[n * 24 + j] = acc * 0.0625f;                      // 1/sqrt(256)
    } else if (idx < 64) {                // zs [MS]
      int o = idx - 48; float acc = 0.f;
      for (int i = 0; i < 16; ++i) {
        float yi = ysL[w][i];
        const float* row = lin1_s + (size_t)((layer * 16 + i) * 32) * 16 + o;
#pragma unroll 8
        for (int e = 0; e < 32; ++e) acc += yi * aL[w][e] * row[e * 16];
      }
      zs[n * 16 + o] = acc * 0.044194174f;
    } else {                              // zv [MV,3]
      int j = idx - 64, o = j / 3, d = j % 3; float acc = 0.f;
      for (int i = 0; i < 8; ++i) {
        float yi = yvL[w][i * 3 + d];
        const float* row = lin1_v + (size_t)((layer * 8 + i) * 32) * 8 + o;
#pragma unroll 8
        for (int e = 0; e < 32; ++e) acc += yi * aL[w][e] * row[e * 8];
      }
      zv[n * 24 + j] = acc * 0.0625f;
    }
  }
}

__global__ void k_zero(float* __restrict__ p, int n) {
  int i = blockIdx.x * blockDim.x + threadIdx.x;
  if (i < n) p[i] = 0.f;
}

// ------------- edge kernel: geometry -> WMMA MLP -> TP -> atomic scatter ---
__global__ __launch_bounds__(64) void k_edge(
    const float* __restrict__ xcur, const int* __restrict__ esrc, const int* __restrict__ edst,
    const float* __restrict__ zsG, const float* __restrict__ zvG,
    const _Float16* __restrict__ W1B, const _Float16* __restrict__ W2B,
    float* __restrict__ m_s, float* __restrict__ m_v, int E) {
  __shared__ __align__(16) _Float16 wLDS[2][16][648];   // per-edge TP weights (f16)
  __shared__ __align__(16) _Float16 hidLDS[2][16][64];  // post-silu hidden (f16)
  __shared__ float featLDS[2][16][16];
  __shared__ float eattrLDS[2][16][6];
  __shared__ float zsLDS[2][16][16];
  __shared__ float zvLDS[2][16][24];

  const int w = threadIdx.x >> 5;
  const int l = threadIdx.x & 31;
  const int base = blockIdx.x * 32 + w * 16;
  const bool active = (base + 16) <= E;   // uniform per wave (E multiple of 16)

  // ---- phase 1: geometry (lanes 0..15) + zs/zv gather (lanes 16..31) ----
  if (active) {
    if (l < 16) {
      int m = l, e = base + m;
      int s = esrc[e], d = edst[e];
      (void)d;
      float xs[6], xd[6];
#pragma unroll
      for (int j = 0; j < 6; ++j) { xs[j] = xcur[s * 6 + j]; xd[j] = xcur[d * 6 + j]; }
#pragma unroll
      for (int c = 0; c < 2; ++c) {
        float e0 = xs[c * 3 + 0] - xd[c * 3 + 0];
        float e1 = xs[c * 3 + 1] - xd[c * 3 + 1];
        float e2 = xs[c * 3 + 2] - xd[c * 3 + 2];
        float d2 = e0 * e0 + e1 * e1 + e2 * e2;
        float inv = __builtin_amdgcn_rsqf(d2);            // 1/ln
        float ln = d2 * inv;                              // sqrt(d2)
        float u = ln - 2.f;                               // 2*(ln/MAX_R - 1), MAX_R=2
        float cut = (u > 0.f) ? 0.f : ((u < -1.f) ? 1.f : 0.5f * (1.f - __cosf(PI_F * u)));
        float cc = cut * 1.7320508f * inv;                // cutoff * sqrt(3)/ln
        eattrLDS[w][m][c * 3 + 0] = cc * e0;
        eattrLDS[w][m][c * 3 + 1] = cc * e1;
        eattrLDS[w][m][c * 3 + 2] = cc * e2;
        float ang = ln * (PI_F * 0.5f);
#pragma unroll
        for (int nb = 1; nb <= NN_NB; ++nb)
          featLDS[w][m][c * 8 + nb - 1] = __sinf((float)nb * ang) * inv * 2.8284271f; // sqrt(NB)
      }
    } else {
      int m = l - 16, e = base + m;
      int s = esrc[e];
#pragma unroll
      for (int j = 0; j < 16; ++j) zsLDS[w][m][j] = zsG[s * 16 + j];
#pragma unroll
      for (int j = 0; j < 24; ++j) zvLDS[w][m][j] = zvG[s * 24 + j];
    }
  }
  __syncthreads();

  const int m16 = l & 15, half = l >> 4, kb = half * 8;
  v8f zeroc = {};

  // ---- phase 2a: hid = silu(feat @ W1 / sqrt(32)) via WMMA (scale in W1B) ----
  if (active) {
    v16h A;
#pragma unroll
    for (int j = 0; j < 8; ++j) {
      A[j] = (_Float16)featLDS[w][m16][kb + j];
      A[8 + j] = (_Float16)0.f;                            // padded K=16..31
    }
#pragma unroll
    for (int t = 0; t < 4; ++t) {
      v16h B = *reinterpret_cast<const v16h*>(&W1B[(t * 32 + l) * 16]);
      v8f d = wmma16x16x32(A, B, zeroc);
#pragma unroll
      for (int r = 0; r < 8; ++r)
        hidLDS[w][r + half * 8][t * 16 + m16] = (_Float16)silu_f(d[r]);
    }
  }
  __syncthreads();

  // ---- phase 2b: w = hid @ W2 / sqrt(64) via WMMA (scale in W2B) ----
  if (active) {
    v8h lo0 = *reinterpret_cast<const v8h*>(&hidLDS[w][m16][kb]);
    v8h hi0 = *reinterpret_cast<const v8h*>(&hidLDS[w][m16][16 + kb]);
    v8h lo1 = *reinterpret_cast<const v8h*>(&hidLDS[w][m16][32 + kb]);
    v8h hi1 = *reinterpret_cast<const v8h*>(&hidLDS[w][m16][48 + kb]);
    v16h A0 = __builtin_shufflevector(lo0, hi0, 0,1,2,3,4,5,6,7,8,9,10,11,12,13,14,15);
    v16h A1 = __builtin_shufflevector(lo1, hi1, 0,1,2,3,4,5,6,7,8,9,10,11,12,13,14,15);
#pragma unroll
    for (int t = 0; t < 40; ++t) {
      v16h B0 = *reinterpret_cast<const v16h*>(&W2B[((t * 2 + 0) * 32 + l) * 16]);
      v16h B1 = *reinterpret_cast<const v16h*>(&W2B[((t * 2 + 1) * 32 + l) * 16]);
      v8f acc = wmma16x16x32(A0, B0, zeroc);
      acc = wmma16x16x32(A1, B1, acc);
#pragma unroll
      for (int r = 0; r < 8; ++r)
        wLDS[w][r + half * 8][t * 16 + m16] = (_Float16)acc[r];
    }
  }
  __syncthreads();

  // ---- phase 3: CG tensor product + atomic scatter (2 lanes per edge) ----
  if (active) {
    int m = l >> 1, role = l & 1, e = base + m;
    int dn = edst[e];
    float ea[6];
#pragma unroll
    for (int j = 0; j < 6; ++j) ea[j] = eattrLDS[w][m][j];
    if (role == 0) {        // out_v[o,d] = sum_{i,c} zs[i]*eattr[c,d]*wA[i,c,o]
      float zsv[16];
#pragma unroll
      for (int i = 0; i < 16; ++i) zsv[i] = zsLDS[w][m][i];
      float acc[8][3];
#pragma unroll
      for (int o = 0; o < 8; ++o)
#pragma unroll
        for (int d = 0; d < 3; ++d) acc[o][d] = 0.f;
      for (int i = 0; i < 16; ++i) {
        float s = zsv[i];
#pragma unroll
        for (int c = 0; c < 2; ++c) {
          v8h wa = *reinterpret_cast<const v8h*>(&wLDS[w][m][i * 16 + c * 8]);
#pragma unroll
          for (int d = 0; d < 3; ++d) {
            float coef = s * ea[c * 3 + d];
#pragma unroll
            for (int o = 0; o < 8; ++o) acc[o][d] += coef * (float)wa[o];
          }
        }
      }
#pragma unroll
      for (int o = 0; o < 8; ++o)
#pragma unroll
        for (int d = 0; d < 3; ++d)
          atomicAdd(&m_v[dn * 24 + o * 3 + d], acc[o][d] * 0.044194174f); // 1/(sqrt(32)*4)
    } else {                // out_s[o] = sum_{i,c,d} zv[i,d]*eattr[c,d]*wB[i,c,o]
      float zvv[24];
#pragma unroll
      for (int j = 0; j < 24; ++j) zvv[j] = zvLDS[w][m][j];
      float accs[24];
#pragma unroll
      for (int o = 0; o < 24; ++o) accs[o] = 0.f;
      for (int i = 0; i < 8; ++i) {
#pragma unroll
        for (int c = 0; c < 2; ++c) {
          float coef = zvv[i * 3 + 0] * ea[c * 3 + 0] + zvv[i * 3 + 1] * ea[c * 3 + 1] +
                       zvv[i * 3 + 2] * ea[c * 3 + 2];
          const _Float16* wb = &wLDS[w][m][NN_NA + i * 48 + c * 24];
#pragma unroll
          for (int o = 0; o < 24; ++o) accs[o] += coef * (float)wb[o];
        }
      }
#pragma unroll
      for (int o = 0; o < 24; ++o)
        atomicAdd(&m_s[dn * 24 + o], accs[o] * 0.036084392f); // 1/(4*sqrt(3)*4)
    }
  }
}

// ------------- node update: lin2, gate, self-interaction, leapfrog, x ------
__global__ __launch_bounds__(256) void k_nodeupd(
    const float* __restrict__ a, const float* __restrict__ scs, const float* __restrict__ scv,
    const float* __restrict__ m_s, const float* __restrict__ m_v,
    const float* __restrict__ lin2_s, const float* __restrict__ lin2_v,
    const float* __restrict__ si_s, const float* __restrict__ si_v,
    const float* __restrict__ Kc, const float* __restrict__ hArr, const float* __restrict__ mixArr,
    float* __restrict__ ys, float* __restrict__ ys_old,
    float* __restrict__ yv, float* __restrict__ yv_old,
    float* __restrict__ xcur, int layer, int Nn) {
  __shared__ float aL[8][32], ysL[8][16], yvL[8][24], msL[8][24], mvL[8][24], csL[8][24], yvN[8][24];
  int w = threadIdx.x >> 5, l = threadIdx.x & 31;
  int n = blockIdx.x * 8 + w;
  bool act = n < Nn;
  if (act) {
    aL[w][l] = a[n * 32 + l];
    if (l < 16) ysL[w][l] = ys[n * 16 + l];
    if (l < 24) { yvL[w][l] = yv[n * 24 + l]; msL[w][l] = m_s[n * 24 + l]; mvL[w][l] = m_v[n * 24 + l]; }
  }
  __syncthreads();
  float cvout = 0.f;
  if (act && l < 24) {
    int o = l; float acc = 0.f;
    for (int i = 0; i < 24; ++i) {
      float mi = msL[w][i];
      const float* row = lin2_s + (size_t)((layer * 24 + i) * 32) * 24 + o;
#pragma unroll 8
      for (int e = 0; e < 32; ++e) acc += mi * aL[w][e] * row[e * 24];
    }
    float o_s = acc * 0.036084392f;                       // 1/sqrt(768)
    csL[w][o] = 0.38268343f * scs[n * 24 + o] + 0.92387953f * o_s;
    int ov = l / 3, dd = l % 3; float acc2 = 0.f;
    for (int i = 0; i < 8; ++i) {
      float mi = mvL[w][i * 3 + dd];
      const float* row = lin2_v + (size_t)((layer * 8 + i) * 32) * 8 + ov;
#pragma unroll 8
      for (int e = 0; e < 32; ++e) acc2 += mi * aL[w][e] * row[e * 8];
    }
    cvout = 0.38268343f * scv[n * 24 + l] + 0.92387953f * (acc2 * 0.0625f);
  }
  __syncthreads();
  if (act) {
    float hv = hArr[layer];
    float dt = fminf(fmaxf(hv * hv, 1e-4f), 0.1f);
    float mx = mixArr[layer];
    float mm = fminf(mx * mx, 1.0f);
    if (l < 16) {
      int j = l;
      float cs = csL[w][j];
      float gs = silu_f(cs);
      float sis = 0.f;
#pragma unroll
      for (int i = 0; i < 16; ++i) sis += ysL[w][i] * si_s[(layer * 16 + i) * 16 + j];
      sis *= 0.25f;                                        // 1/sqrt(MS)
      float dsv = mm * gs + (1.f - mm) * sis;
      float cur = ysL[w][j];
      float neu = 2.f * cur - ys_old[n * 16 + j] + dt * dsv;
      ys_old[n * 16 + j] = cur;
      ys[n * 16 + j] = neu;
    }
    if (l < 24) {
      int o = l / 3, dd = l % 3;
      float g = sigm_f(csL[w][16 + o]);
      float gv = g * cvout;
      float siv = 0.f;
#pragma unroll
      for (int i = 0; i < 8; ++i) siv += yvL[w][i * 3 + dd] * si_v[(layer * 8 + i) * 8 + o];
      siv *= 0.35355339f;                                  // 1/sqrt(MV)
      float dv = mm * gv + (1.f - mm) * siv;
      float cur = yvL[w][l];
      float neu = 2.f * cur - yv_old[n * 24 + l] + dt * dv;
      yv_old[n * 24 + l] = cur;
      yv[n * 24 + l] = neu;
      yvN[w][l] = neu;
    }
  }
  __syncthreads();
  if (act && l < 6) {
    int c = l / 3, dd = l % 3; float acc = 0.f;
#pragma unroll
    for (int k = 0; k < 8; ++k) acc += yvN[w][k * 3 + dd] * Kc[c * 8 + k];
    xcur[n * 6 + l] = acc;
  }
}

__global__ void k_final(const float* __restrict__ xcur, float* __restrict__ out, int Nn) {
  int i = blockIdx.x * blockDim.x + threadIdx.x;
  int tot = Nn * 6;
  if (i < tot) out[i] = xcur[i];
  else if (i == tot)     out[i] = -1.f;
  else if (i == tot + 1) out[i] = -1.f;
  else if (i == tot + 2) out[i] = 0.f;
}

// ------------------------------- launch -----------------------------------
extern "C" void kernel_launch(void* const* d_in, const int* in_sizes, int n_in,
                              void* d_out, int out_size, void* d_ws, size_t ws_size,
                              hipStream_t stream) {
  const float* x        = (const float*)d_in[0];
  const int*   node_attr= (const int*)d_in[2];
  const int*   esrc     = (const int*)d_in[3];
  const int*   edst     = (const int*)d_in[4];
  const float* emb      = (const float*)d_in[5];
  const float* Kc       = (const float*)d_in[6];
  const float* hArr     = (const float*)d_in[7];
  const float* mixArr   = (const float*)d_in[8];
  const float* sc_s     = (const float*)d_in[9];
  const float* sc_v     = (const float*)d_in[10];
  const float* lin1_s   = (const float*)d_in[11];
  const float* lin1_v   = (const float*)d_in[12];
  const float* fc_w1    = (const float*)d_in[13];
  const float* fc_w2    = (const float*)d_in[14];
  const float* lin2_s   = (const float*)d_in[15];
  const float* lin2_v   = (const float*)d_in[16];
  const float* si_s     = (const float*)d_in[17];
  const float* si_v     = (const float*)d_in[18];

  const int Nn = in_sizes[0] / 6;
  const int E  = in_sizes[3];

  char* ws = (char*)d_ws;
  size_t off = 0;
  auto alloc = [&](size_t bytes) -> void* {
    void* p = ws + off;
    off += (bytes + 255) & ~(size_t)255;
    return p;
  };
  float* a       = (float*)alloc((size_t)Nn * 32 * 4);
  float* ys      = (float*)alloc((size_t)Nn * 16 * 4);
  float* ys_old  = (float*)alloc((size_t)Nn * 16 * 4);
  float* yv      = (float*)alloc((size_t)Nn * 24 * 4);
  float* yv_old  = (float*)alloc((size_t)Nn * 24 * 4);
  float* xcur    = (float*)alloc((size_t)Nn * 6 * 4);
  float* zs      = (float*)alloc((size_t)Nn * 16 * 4);
  float* zv      = (float*)alloc((size_t)Nn * 24 * 4);
  float* scs     = (float*)alloc((size_t)Nn * 24 * 4);
  float* scv     = (float*)alloc((size_t)Nn * 24 * 4);
  float* m_s     = (float*)alloc((size_t)Nn * 48 * 4);   // m_s (24) then m_v (24), contiguous
  float* m_v     = m_s + (size_t)Nn * 24;
  _Float16* W1B  = (_Float16*)alloc((size_t)NN_L * 4 * 32 * 16 * 2);
  _Float16* W2B  = (_Float16*)alloc((size_t)NN_L * 40 * 2 * 32 * 16 * 2);

  k_convw1<<<(NN_L * 4 * 32 * 16 + 255) / 256, 256, 0, stream>>>(fc_w1, W1B);
  k_convw2<<<(NN_L * 40 * 2 * 32 * 16 + 255) / 256, 256, 0, stream>>>(fc_w2, W2B);
  k_init<<<(Nn + 255) / 256, 256, 0, stream>>>(x, node_attr, emb, Kc, a, ys, ys_old, yv, yv_old,
                                               xcur, Nn);
  for (int layer = 0; layer < NN_L; ++layer) {
    k_nodelin<<<(Nn + 7) / 8, 256, 0, stream>>>(a, ys, yv, sc_s, sc_v, lin1_s, lin1_v,
                                                scs, scv, zs, zv, layer, Nn);
    k_zero<<<(Nn * 48 + 255) / 256, 256, 0, stream>>>(m_s, Nn * 48);
    k_edge<<<(E + 31) / 32, 64, 0, stream>>>(xcur, esrc, edst, zs, zv,
                                             W1B + (size_t)layer * 4 * 32 * 16,
                                             W2B + (size_t)layer * 40 * 2 * 32 * 16,
                                             m_s, m_v, E);
    k_nodeupd<<<(Nn + 7) / 8, 256, 0, stream>>>(a, scs, scv, m_s, m_v, lin2_s, lin2_v,
                                                si_s, si_v, Kc, hArr, mixArr,
                                                ys, ys_old, yv, yv_old, xcur, layer, Nn);
  }
  k_final<<<(Nn * 6 + 3 + 255) / 256, 256, 0, stream>>>(xcur, (float*)d_out, Nn);
}
